// HGCL_52458730553693
// MI455X (gfx1250) — matrix operand order repaired and checked by reading, not compiled, not run on hardware
//
#include <hip/hip_runtime.h>
#include <math.h>

#define D 128
#define JSTRIP 32   // j-tiles per wave in SSL GEMM

typedef __attribute__((ext_vector_type(16))) __bf16 v16bf;
typedef __attribute__((ext_vector_type(8)))  __bf16 v8bf;
typedef __attribute__((ext_vector_type(8)))  float  v8f;

// ---------- helpers ----------
__device__ __forceinline__ v16bf cat8(v8bf a, v8bf b) {
  return __builtin_shufflevector(a, b, 0, 1, 2, 3, 4, 5, 6, 7, 8, 9, 10, 11, 12, 13, 14, 15);
}

// A fragment (16x32) from f32 row-major [rows x 128], native v_cvt_pk_bf16_f32.
// lane L(<16): row M=L, K = {k0..k0+7} U {k0+16..k0+23}; lanes>=16 shifted by 8.
__device__ __forceinline__ v16bf load_fragA_f32(const float* base, int row0, int k0, int lane) {
  const float* r = base + (long)(row0 + (lane & 15)) * D + k0 + ((lane >> 4) << 3);
  v16bf f;
#pragma unroll
  for (int e = 0; e < 8; ++e) f[e] = (__bf16)r[e];
#pragma unroll
  for (int e = 0; e < 8; ++e) f[8 + e] = (__bf16)r[16 + e];
  return f;
}

// A fragment from bf16 row-major [rows x 128]: two aligned b128 loads.
__device__ __forceinline__ v16bf load_fragA_bf(const __bf16* base, int row0, int k0, int lane) {
  const __bf16* r = base + (long)(row0 + (lane & 15)) * D + k0 + ((lane >> 4) << 3);
  v8bf lo = *(const v8bf*)(r);
  v8bf hi = *(const v8bf*)(r + 16);
  return cat8(lo, hi);
}

// B fragment (32x16) from bf16 row-major B^T [cols x 128]:
// lane L(<16): col N=L, K = {k0..k0+7} U {k0+16..k0+23}; lanes>=16 shifted by 8.
__device__ __forceinline__ v16bf load_fragBt_bf(const __bf16* base, int n0, int k0, int lane) {
  const __bf16* r = base + (long)(n0 + (lane & 15)) * D + k0 + ((lane >> 4) << 3);
  v8bf lo = *(const v8bf*)(r);
  v8bf hi = *(const v8bf*)(r + 16);
  return cat8(lo, hi);
}

__device__ __forceinline__ v8f wmma_bf16(v16bf a, v16bf b, v8f c) {
  return __builtin_amdgcn_wmma_f32_16x16x32_bf16(false, a, false, b, (short)0, c, false, false);
}

__device__ __forceinline__ float waveSum(float v) {
#pragma unroll
  for (int o = 16; o > 0; o >>= 1) v += __shfl_xor(v, o, 32);
  return v;
}

// branch-free tanh: 1 - 2/(e^{2x}+1); correct limits at +-inf, no exec divergence
__device__ __forceinline__ float fast_tanh(float x) {
  float ex = __expf(2.f * x);
  return 1.f - 2.f * __builtin_amdgcn_rcpf(ex + 1.f);
}

// ---------- elementwise / scatter kernels ----------
__global__ void k_zero(float* p, long n) {
  long i = (long)blockIdx.x * blockDim.x + threadIdx.x;
  long s = (long)gridDim.x * blockDim.x;
  for (; i < n; i += s) p[i] = 0.f;
}

__global__ void k_copy(float* dst, const float* src, long n) {
  long i = (long)blockIdx.x * blockDim.x + threadIdx.x;
  long s = (long)gridDim.x * blockDim.x;
  for (; i < n; i += s) dst[i] = src[i];
}

__global__ void k_setv(float* p, float v) { *p = v; }

// transpose-convert a 128x128 f32 matrix into bf16 (stores W^T row-major)
__global__ void k_t128_bf(const float* in, __bf16* out) {
  int i = blockIdx.x * blockDim.x + threadIdx.x;
  int k = i >> 7, n = i & 127;
  out[(long)n * D + k] = (__bf16)in[(long)k * D + n];
}

__global__ void k_deg(float* deg, const int* idx, int E) {
  long i = (long)blockIdx.x * blockDim.x + threadIdx.x;
  long s = (long)gridDim.x * blockDim.x;
  for (; i < E; i += s) atomicAdd(&deg[idx[i]], 1.f);
}

__global__ void k_rsqrt1(float* out, const float* deg, int n) {
  long i = (long)blockIdx.x * blockDim.x + threadIdx.x;
  long s = (long)gridDim.x * blockDim.x;
  for (; i < n; i += s) out[i] = rsqrtf(fmaxf(deg[i], 1.f));
}

__global__ void k_dinvk(float* out, const float* deg, int n) {
  long i = (long)blockIdx.x * blockDim.x + threadIdx.x;
  long s = (long)gridDim.x * blockDim.x;
  for (; i < n; i += s) { float d = deg[i]; out[i] = d > 0.f ? rsqrtf(d) : 0.f; }
}

__global__ void k_conv_scatter(const float* __restrict__ h, const int* __restrict__ src,
                               const int* __restrict__ dst, const float* __restrict__ rso,
                               float* agg, long total) {
  long i = (long)blockIdx.x * blockDim.x + threadIdx.x;
  long st = (long)gridDim.x * blockDim.x;
  for (; i < total; i += st) {
    long e = i >> 7; int d = (int)(i & 127);
    int s = src[e], t = dst[e];
    atomicAdd(&agg[(long)t * D + d], h[(long)s * D + d] * rso[s]);
  }
}

__global__ void k_scale_rows(float* a, const float* rsi, long total) {
  long i = (long)blockIdx.x * blockDim.x + threadIdx.x;
  long s = (long)gridDim.x * blockDim.x;
  for (; i < total; i += s) a[i] *= rsi[i >> 7];
}

__global__ void k_ui_scatter(const float* __restrict__ x, const int* __restrict__ row,
                             const int* __restrict__ col, const float* __restrict__ dinv,
                             float* xn, long total) {
  long i = (long)blockIdx.x * blockDim.x + threadIdx.x;
  long st = (long)gridDim.x * blockDim.x;
  for (; i < total; i += st) {
    long e = i >> 7; int d = (int)(i & 127);
    int r = row[e], c = col[e];
    atomicAdd(&xn[(long)r * D + d], dinv[r] * dinv[c] * x[(long)c * D + d]);
  }
}

__global__ void k_beta(float* wsum, float inv_n) {
  float m0 = wsum[0] * inv_n, m1 = wsum[1] * inv_n;
  float mx = fmaxf(m0, m1);
  float e0 = __expf(m0 - mx), e1 = __expf(m1 - mx);
  float inv = 1.f / (e0 + e1);
  wsum[2] = e0 * inv; wsum[3] = e1 * inv;
}

__global__ void k_combine(float* h, const float* z0, const float* z1, const float* beta, long n) {
  float b0 = beta[2], b1v = beta[3];
  long i = (long)blockIdx.x * blockDim.x + threadIdx.x;
  long s = (long)gridDim.x * blockDim.x;
  for (; i < n; i += s) h[i] = b0 * z0[i] + b1v * z1[i];
}

__global__ void k_halfsum(float* o, const float* a, const float* b, long n) {
  long i = (long)blockIdx.x * blockDim.x + threadIdx.x;
  long s = (long)gridDim.x * blockDim.x;
  for (; i < n; i += s) o[i] = 0.5f * (a[i] + b[i]);
}

__global__ void k_gather(const float* src, const int* idx, float* out, long total) {
  long i = (long)blockIdx.x * blockDim.x + threadIdx.x;
  long s = (long)gridDim.x * blockDim.x;
  for (; i < total; i += s) {
    long b = i >> 7; int d = (int)(i & 127);
    out[i] = src[(long)idx[b] * D + d];
  }
}

// gather + L2-normalize; emit both f32 (for the exact diagonal dot) and bf16 (for WMMA)
__global__ void k_gather_norm(const float* src, const int* idx, float* out, __bf16* outb) {
  __shared__ float red[4];
  long b = blockIdx.x; int d = threadIdx.x;
  float v = src[(long)idx[b] * D + d];
  float s = waveSum(v * v);
  if ((d & 31) == 0) red[d >> 5] = s;
  __syncthreads();
  float tot = red[0] + red[1] + red[2] + red[3];
  float inv = 1.f / fmaxf(sqrtf(tot), 1e-12f);
  float nv = v * inv;
  out[b * D + d] = nv;
  outb[b * D + d] = (__bf16)nv;
}

// ---------- WMMA kernels ----------
// semantic attention accumulator: wsum += sum_rows( tanh(z@W1+b1) @ W2 )
__global__ void k_semw(const float* __restrict__ z, const __bf16* __restrict__ W1t,
                       const float* __restrict__ b1, const float* __restrict__ W2,
                       float* wsum, int ntiles) {
  int lane = threadIdx.x & 31;
  int tile = blockIdx.x * (blockDim.x >> 5) + (threadIdx.x >> 5);
  if (tile >= ntiles) return;                 // wave-uniform: EXEC all-1 at WMMA
  int row0 = tile << 4;
  v16bf aF[4];
#pragma unroll
  for (int kt = 0; kt < 4; ++kt) aF[kt] = load_fragA_f32(z, row0, kt * 32, lane);
  float partial = 0.f;
#pragma unroll
  for (int nt = 0; nt < 8; ++nt) {
    v8f acc = {};
#pragma unroll
    for (int kt = 0; kt < 4; ++kt)
      acc = wmma_bf16(aF[kt], load_fragBt_bf(W1t, nt * 16, kt * 32, lane), acc);
    int colc = nt * 16 + (lane & 15);
    float w2c = W2[colc], bc = b1[colc];
#pragma unroll
    for (int r = 0; r < 8; ++r) partial += fast_tanh(acc[r] + bc) * w2c;
  }
  partial = waveSum(partial);
  if (lane == 0) atomicAdd(wsum, partial);
}

// alls[i] += sum_j exp(2 * n1_i . n2_j); wave keeps A resident, sweeps JSTRIP j-tiles.
// Per-lane exp partials accumulate in registers; single cross-lane reduce at the end.
__global__ void k_ssl_gemm(const __bf16* __restrict__ n1b, const __bf16* __restrict__ n2b,
                           float* alls) {
  int lane = threadIdx.x & 31;
  int wave = threadIdx.x >> 5;
  int it = blockIdx.y * 4 + wave;
  int i0 = it << 4;
  v16bf aF[4];
#pragma unroll
  for (int kt = 0; kt < 4; ++kt) aF[kt] = load_fragA_bf(n1b, i0, kt * 32, lane);
  float rsum[8];
#pragma unroll
  for (int r = 0; r < 8; ++r) rsum[r] = 0.f;
  int jt0 = blockIdx.x * JSTRIP;
  for (int js = 0; js < JSTRIP; ++js) {
    int j0 = (jt0 + js) << 4;
    if (js + 1 < JSTRIP) {  // prefetch next B tile row (global_prefetch_b8)
      const __bf16* nxt = n2b + (long)(j0 + 16 + (lane & 15)) * D;
      __builtin_prefetch(nxt, 0, 1);
    }
    v8f acc = {};
#pragma unroll
    for (int kt = 0; kt < 4; ++kt)
      acc = wmma_bf16(aF[kt], load_fragBt_bf(n2b, j0, kt * 32, lane), acc);
#pragma unroll
    for (int r = 0; r < 8; ++r) rsum[r] += __expf(2.f * acc[r]);
  }
#pragma unroll
  for (int r = 0; r < 8; ++r) {
    float p = rsum[r];
#pragma unroll
    for (int m = 1; m <= 8; m <<= 1) p += __shfl_xor(p, m, 32);  // one 16-col reduce
    rsum[r] = p;
  }
  if ((lane & 15) == 0) {
    int rbase = i0 + ((lane >> 4) << 3);
#pragma unroll
    for (int r = 0; r < 8; ++r) atomicAdd(&alls[rbase + r], rsum[r]);
  }
}

__global__ void k_ssl_loss(const float* n1, const float* n2, const float* alls,
                           float* loss, float coef) {
  __shared__ float red[4];
  long b = blockIdx.x; int d = threadIdx.x;
  float s = waveSum(n1[b * D + d] * n2[b * D + d]);
  if ((d & 31) == 0) red[d >> 5] = s;
  __syncthreads();
  if (d == 0) {
    float dot = red[0] + red[1] + red[2] + red[3];
    atomicAdd(loss, coef * -(2.f * dot - __logf(alls[b])));
  }
}

// Y = relu(X @ W + bias); W given pre-transposed in bf16 (Wt[n][k])
__global__ void k_gemm_bias_relu(const float* __restrict__ X, const __bf16* __restrict__ Wt,
                                 const float* __restrict__ bias, float* Y, int ntiles) {
  int lane = threadIdx.x & 31;
  int tile = blockIdx.x * (blockDim.x >> 5) + (threadIdx.x >> 5);
  if (tile >= ntiles) return;
  int row0 = tile << 4;
  v16bf aF[4];
#pragma unroll
  for (int kt = 0; kt < 4; ++kt) aF[kt] = load_fragA_f32(X, row0, kt * 32, lane);
#pragma unroll
  for (int nt = 0; nt < 8; ++nt) {
    v8f acc = {};
#pragma unroll
    for (int kt = 0; kt < 4; ++kt)
      acc = wmma_bf16(aF[kt], load_fragBt_bf(Wt, nt * 16, kt * 32, lane), acc);
    int colc = nt * 16 + (lane & 15);
    float bb = bias[colc];
    int rb = row0 + ((lane >> 4) << 3);
#pragma unroll
    for (int r = 0; r < 8; ++r)
      Y[(long)(rb + r) * D + colc] = fmaxf(acc[r] + bb, 0.f);
  }
}

__global__ void k_layernorm(const float* Y, const float* g, const float* b, float* out) {
  __shared__ float red[4];
  long row = blockIdx.x; int d = threadIdx.x;
  float v = Y[row * D + d];
  float s = waveSum(v);
  if ((d & 31) == 0) red[d >> 5] = s;
  __syncthreads();
  float mu = (red[0] + red[1] + red[2] + red[3]) * (1.f / D);
  float c = v - mu;
  __syncthreads();
  float s2 = waveSum(c * c);
  if ((d & 31) == 0) red[d >> 5] = s2;
  __syncthreads();
  float var = (red[0] + red[1] + red[2] + red[3]) * (1.f / D);
  out[row * D + d] = c * rsqrtf(var + 1e-5f) * g[d] + b[d];
}

// ---------- orchestration ----------
static unsigned nb(long n, int t) {
  long b = (n + t - 1) / t;
  if (b < 1) b = 1;
  if (b > 1048576) b = 1048576;
  return (unsigned)b;
}

extern "C" void kernel_launch(void* const* d_in, const int* in_sizes, int n_in,
                              void* d_out, int out_size, void* d_ws, size_t ws_size,
                              hipStream_t stream) {
  (void)n_in; (void)ws_size;
  const float* feat_u = (const float*)d_in[0];
  const float* feat_i = (const float*)d_in[1];
  const float* uW1 = (const float*)d_in[2];
  const float* ub1 = (const float*)d_in[3];
  const float* uW2 = (const float*)d_in[4];
  const float* iW1 = (const float*)d_in[5];
  const float* ib1 = (const float*)d_in[6];
  const float* iW2 = (const float*)d_in[7];
  const float* userW = (const float*)d_in[8];
  const float* userb = (const float*)d_in[9];
  const float* itemW = (const float*)d_in[10];
  const float* itemb = (const float*)d_in[11];
  const float* lng = (const float*)d_in[12];
  const float* lnb = (const float*)d_in[13];
  const int* ump0s = (const int*)d_in[14];
  const int* ump0d = (const int*)d_in[15];
  const int* ump1s = (const int*)d_in[16];
  const int* ump1d = (const int*)d_in[17];
  const int* imp0s = (const int*)d_in[18];
  const int* imp0d = (const int*)d_in[19];
  const int* imp1s = (const int*)d_in[20];
  const int* imp1d = (const int*)d_in[21];
  const int* uirow = (const int*)d_in[22];
  const int* uicol = (const int*)d_in[23];
  const int* uidx = (const int*)d_in[24];
  const int* iidx = (const int*)d_in[25];
  const int* nidx = (const int*)d_in[26];

  const int NU = in_sizes[0] / D, NI = in_sizes[1] / D, N = NU + NI;
  const int Eui = in_sizes[22];
  const int B = in_sizes[24];
  const long ND_u = (long)NU * D, ND_i = (long)NI * D, NDn = (long)N * D, BD = (long)B * D;
  const int M = NU > NI ? NU : NI;
  const long ND_m = (long)M * D;

  float* ws = (float*)d_ws;
  float* h_u = ws;    ws += ND_u;
  float* h_i = ws;    ws += ND_i;
  float* z0 = ws;     ws += ND_m;
  float* z1 = ws;     ws += ND_m;
  float* x  = ws;     ws += NDn;
  float* xn = ws;     ws += NDn;
  float* emb_u = ws;  ws += ND_u;
  float* emb_i = ws;  ws += ND_i;
  float* n1 = ws;     ws += BD;
  float* n2 = ws;     ws += BD;
  float* yb = ws;     ws += BD;
  float* rs = ws;     ws += 8L * M;
  float* dinv = ws;   ws += N;
  float* degt = ws;   ws += N;   // also reused as `alls` (B <= N)
  float* wsum = ws;   ws += 8;
  __bf16* uW1t  = (__bf16*)ws; ws += 8192;   // 128*128 bf16
  __bf16* iW1t  = (__bf16*)ws; ws += 8192;
  __bf16* userWt = (__bf16*)ws; ws += 8192;
  __bf16* itemWt = (__bf16*)ws; ws += 8192;
  __bf16* n1b = (__bf16*)ws; ws += BD / 2;
  __bf16* n2b = (__bf16*)ws; ws += BD / 2;

  float* rs_u0o = rs + 0L * M; float* rs_u0i = rs + 1L * M;
  float* rs_u1o = rs + 2L * M; float* rs_u1i = rs + 3L * M;
  float* rs_i0o = rs + 4L * M; float* rs_i0i = rs + 5L * M;
  float* rs_i1o = rs + 6L * M; float* rs_i1i = rs + 7L * M;

  // one-time bf16 transposes of the dense weights
  k_t128_bf<<<64, 256, 0, stream>>>(uW1, uW1t);
  k_t128_bf<<<64, 256, 0, stream>>>(iW1, iW1t);
  k_t128_bf<<<64, 256, 0, stream>>>(userW, userWt);
  k_t128_bf<<<64, 256, 0, stream>>>(itemW, itemWt);

  auto calc_rs = [&](const int* idx, int E, int n, float* out) {
    k_zero<<<nb(n, 256), 256, 0, stream>>>(degt, n);
    k_deg<<<nb(E, 256), 256, 0, stream>>>(degt, idx, E);
    k_rsqrt1<<<nb(n, 256), 256, 0, stream>>>(out, degt, n);
  };
  calc_rs(ump0s, in_sizes[14], NU, rs_u0o); calc_rs(ump0d, in_sizes[15], NU, rs_u0i);
  calc_rs(ump1s, in_sizes[16], NU, rs_u1o); calc_rs(ump1d, in_sizes[17], NU, rs_u1i);
  calc_rs(imp0s, in_sizes[18], NI, rs_i0o); calc_rs(imp0d, in_sizes[19], NI, rs_i0i);
  calc_rs(imp1s, in_sizes[20], NI, rs_i1o); calc_rs(imp1d, in_sizes[21], NI, rs_i1i);

  // bipartite normalization
  k_zero<<<nb(N, 256), 256, 0, stream>>>(degt, N);
  k_deg<<<nb(Eui, 256), 256, 0, stream>>>(degt, uirow, Eui);
  k_dinvk<<<nb(N, 256), 256, 0, stream>>>(dinv, degt, N);

  // 3-layer HAN per graph (graph-conv scatter + WMMA semantic attention)
  auto han = [&](float* h, const float* feat,
                 const int* s0, const int* d0, const float* r0o, const float* r0i, int E0,
                 const int* s1, const int* d1, const float* r1o, const float* r1i, int E1,
                 const __bf16* W1t, const float* b1, const float* W2, int n) {
    long nd = (long)n * D;
    int ntiles = n >> 4;
    k_copy<<<nb(nd, 256), 256, 0, stream>>>(h, feat, nd);
    for (int it = 0; it < 3; ++it) {
      k_zero<<<nb(nd, 256), 256, 0, stream>>>(z0, nd);
      k_zero<<<nb(nd, 256), 256, 0, stream>>>(z1, nd);
      k_conv_scatter<<<nb((long)E0 * D, 256), 256, 0, stream>>>(h, s0, d0, r0o, z0, (long)E0 * D);
      k_scale_rows<<<nb(nd, 256), 256, 0, stream>>>(z0, r0i, nd);
      k_conv_scatter<<<nb((long)E1 * D, 256), 256, 0, stream>>>(h, s1, d1, r1o, z1, (long)E1 * D);
      k_scale_rows<<<nb(nd, 256), 256, 0, stream>>>(z1, r1i, nd);
      k_zero<<<1, 32, 0, stream>>>(wsum, 2);
      k_semw<<<(ntiles + 7) / 8, 256, 0, stream>>>(z0, W1t, b1, W2, wsum + 0, ntiles);
      k_semw<<<(ntiles + 7) / 8, 256, 0, stream>>>(z1, W1t, b1, W2, wsum + 1, ntiles);
      k_beta<<<1, 1, 0, stream>>>(wsum, 1.f / (float)n);
      k_combine<<<nb(nd, 256), 256, 0, stream>>>(h, z0, z1, wsum, nd);
    }
  };
  han(h_u, feat_u, ump0s, ump0d, rs_u0o, rs_u0i, in_sizes[14],
      ump1s, ump1d, rs_u1o, rs_u1i, in_sizes[16], uW1t, ub1, uW2, NU);
  han(h_i, feat_i, imp0s, imp0d, rs_i0o, rs_i0i, in_sizes[18],
      imp1s, imp1d, rs_i1o, rs_i1i, in_sizes[20], iW1t, ib1, iW2, NI);

  // 3-layer LightGCN propagation on user-item graph
  k_copy<<<nb(ND_u, 256), 256, 0, stream>>>(x, feat_u, ND_u);
  k_copy<<<nb(ND_i, 256), 256, 0, stream>>>(x + ND_u, feat_i, ND_i);
  float* xa = x; float* xb = xn;
  for (int it = 0; it < 3; ++it) {
    k_zero<<<nb(NDn, 256), 256, 0, stream>>>(xb, NDn);
    k_ui_scatter<<<nb((long)Eui * D, 256), 256, 0, stream>>>(xa, uirow, uicol, dinv, xb, (long)Eui * D);
    float* t = xa; xa = xb; xb = t;
  }
  k_halfsum<<<nb(ND_u, 256), 256, 0, stream>>>(emb_u, h_u, xa, ND_u);
  k_halfsum<<<nb(ND_i, 256), 256, 0, stream>>>(emb_i, h_i, xa + ND_u, ND_i);

  // SSL losses (WMMA 8192x8192x128 similarity + row softmax denominator)
  float* loss = (float*)d_out + (out_size - 1);
  k_setv<<<1, 1, 0, stream>>>(loss, 0.f);
  auto ssl = [&](const float* dui, const float* demb, const int* idx) {
    k_gather_norm<<<B, 128, 0, stream>>>(dui, idx, n1, n1b);
    k_gather_norm<<<B, 128, 0, stream>>>(demb, idx, n2, n2b);
    k_zero<<<nb(B, 256), 256, 0, stream>>>(degt, B);
    dim3 g((unsigned)((B >> 4) / JSTRIP), (unsigned)(B >> 6));
    k_ssl_gemm<<<g, 128, 0, stream>>>(n1b, n2b, degt);
    k_ssl_loss<<<B, 128, 0, stream>>>(n1, n2, degt, loss, 0.4f / (float)B);
  };
  ssl(xa, emb_u, uidx);
  ssl(xa + ND_u, emb_i, iidx);

  // Output heads: gather only needed rows, WMMA dense + relu, layernorm
  float* out = (float*)d_out;
  int btiles = B >> 4;
  auto head = [&](const float* emb, const int* idx, const __bf16* Wt, const float* bias, float* o) {
    k_gather<<<nb(BD, 256), 256, 0, stream>>>(emb, idx, n1, BD);
    k_gemm_bias_relu<<<(btiles + 7) / 8, 256, 0, stream>>>(n1, Wt, bias, yb, btiles);
    k_layernorm<<<B, 128, 0, stream>>>(yb, lng, lnb, o);
  };
  head(emb_u, uidx, userWt, userb, out);
  head(emb_i, iidx, itemWt, itemb, out + BD);
  head(emb_i, nidx, itemWt, itemb, out + 2 * BD);
}
